// MeanAggregator_36275293782334
// MI455X (gfx1250) — matrix operand order repaired and checked by reading, not compiled, not run on hardware
//
#include <hip/hip_runtime.h>
#include <stdint.h>

// ---------------------------------------------------------------------------
// MI455X (gfx1250) implementation: bf16 WMMA (f32 accumulate) fused
// edge-GEMM + segment-mean, weights pre-packed into WMMA B-fragment layout.
// ---------------------------------------------------------------------------

typedef __attribute__((ext_vector_type(16))) __bf16 v16bf;
typedef __attribute__((ext_vector_type(8)))  float  v8f;

union FragU { v16bf v; uint4 q[2]; unsigned short s[16]; };

#define HDIM   128
#define SEQL   10
#define NT     8        // 128 output cols = 8 N-tiles of 16
#define KS3    12       // 384 / 32
#define KS4    8        // 256 / 32
#define ASTRIDE 392     // 384 + 8 pad (bf16 elems) -> conflict-free ds reads
#define ATILE_BYTES (16 * ASTRIDE * 2)                  // 12544 B per wave
#define WB3_USH (NT * KS3 * 32 * 16)                    // 49152 ushorts (96 KB)
#define WB4_USH (NT * KS4 * 32 * 16)                    // 32768 ushorts (64 KB)
#define WPACK_USH (WB3_USH + WB4_USH)                   // 81920 ushorts
#define WLDS_BYTES (WPACK_USH * 2)                      // 163840 B
#define SMEM_BYTES (WLDS_BYTES + 8 * ATILE_BYTES)       // 264192 B (< 320 KB)

__device__ __forceinline__ unsigned int f2bf(float f) {
  unsigned int u = __float_as_uint(f);
  return (u + 0x7FFFu + ((u >> 16) & 1u)) >> 16;   // round-to-nearest-even
}

__device__ __forceinline__ int lbound(const int* a, int n, int key) {
  int lo = 0, hi = n;
  while (lo < hi) { int mid = (lo + hi) >> 1; if (a[mid] < key) lo = mid + 1; else hi = mid; }
  return lo;
}

// ---------------------------------------------------------------------------
// Kernel 1: pack W3 (128x384) and W4 (128x256) f32 weights into bf16 WMMA
// B-fragment layout:  frag[(nt*KS + ks)][lane][j]  with
//   n = nt*16 + (lane&15),  k = ks*32 + (lane>=16 ? 16 : 0) + j
// ---------------------------------------------------------------------------
__global__ void pack_weights_kernel(const float* __restrict__ W3w,
                                    const float* __restrict__ W4w,
                                    unsigned short* __restrict__ Bpack) {
  int id = blockIdx.x * 256 + threadIdx.x;
  if (id >= WPACK_USH) return;
  unsigned short v;
  if (id < WB3_USH) {
    int j = id & 15; int lane = (id >> 4) & 31; int fg = id >> 9; // nt*KS3+ks
    int nt = fg / KS3, ks = fg - nt * KS3;
    int n = nt * 16 + (lane & 15);
    int k = ks * 32 + ((lane >> 4) << 4) + j;
    v = (unsigned short)f2bf(W3w[n * 384 + k]);
  } else {
    int id2 = id - WB3_USH;
    int j = id2 & 15; int lane = (id2 >> 4) & 31; int fg = id2 >> 9; // nt*KS4+ks
    int nt = fg / KS4, ks = fg - nt * KS4;
    int n = nt * 16 + (lane & 15);
    int k = ks * 32 + ((lane >> 4) << 4) + j;
    v = (unsigned short)f2bf(W4w[n * 256 + k]);
  }
  Bpack[id] = v;
}

// ---------------------------------------------------------------------------
// Kernel 2: broadcast thirds of both outputs.
//   out layout: [ s_embed_seq (T x 384) | att_embed_seq (T x 384) ]
// ---------------------------------------------------------------------------
__global__ void fill_outputs_kernel(const float* __restrict__ self_att,
                                    const int* __restrict__ s_ids,
                                    const int* __restrict__ r_ids,
                                    const float* __restrict__ ent,
                                    const float* __restrict__ rel,
                                    const float* __restrict__ W1w,
                                    const float* __restrict__ W1b,
                                    float* __restrict__ out, int T) {
  int id = blockIdx.x * 256 + threadIdx.x;
  if (id >= T * HDIM) return;
  int h = id & (HDIM - 1);
  int t = id >> 7;
  int b = t / SEQL;
  float se = ent[(size_t)s_ids[b] * HDIM + h];
  float sr = rel[(size_t)r_ids[b] * HDIM + h];
  float sa = fmaxf(self_att[t] * W1w[h] + W1b[h], 0.0f);
  size_t base = (size_t)t * 384;
  size_t O2 = (size_t)T * 384;
  out[base + h]            = se;   // s_embed: self_e
  out[base + 128 + h]      = sr;   // s_embed: self_r
  out[O2 + base + h]       = sa;   // att_embed: self_att_emb
  out[O2 + base + 128 + h] = se;   // att_embed: self_e
}

// ---------------------------------------------------------------------------
// Kernel 3: fused edge-GEMM (bf16 WMMA, f32 acc) + per-edge ReLU + segment
// mean. One wave per segment (sorted seg_ids -> binary-search range, no
// atomics). 8 waves/block share LDS-resident packed weights.
// LDS (264 KB) limits us to 1 block/WGP = 2 waves/SIMD, so let the compiler
// budget a large VGPR count (no spills) via waves_per_eu.
// ---------------------------------------------------------------------------
__global__ void __launch_bounds__(256)
__attribute__((amdgpu_waves_per_eu(2)))
seg_gemm_kernel(const int* __restrict__ nbr_ent,
                const int* __restrict__ nbr_rel,
                const float* __restrict__ nbr_att,
                const int* __restrict__ seg_ids,
                const float* __restrict__ ent,
                const float* __restrict__ rel,
                const float* __restrict__ W1w,
                const float* __restrict__ W1b,
                const float* __restrict__ W3b,
                const float* __restrict__ W4b,
                const unsigned short* __restrict__ Bpack,
                float* __restrict__ out, int E, int T) {
  extern __shared__ char smem[];
  unsigned short* sB3 = (unsigned short*)smem;                    // 96 KB
  unsigned short* sB4 = (unsigned short*)(smem + WB3_USH * 2);    // 64 KB

  int tid = threadIdx.x;
  // Stage packed weights to LDS (163840 B = 10240 uint4).
  {
    uint4* d = (uint4*)smem;
    const uint4* g = (const uint4*)Bpack;
    for (int i = tid; i < WLDS_BYTES / 16; i += 256) d[i] = g[i];
  }
  __syncthreads();

  int lane = tid & 31;
  int wave = tid >> 5;
  unsigned short* Atile = (unsigned short*)(smem + WLDS_BYTES + wave * ATILE_BYTES);

  int col = lane & 15;
  int mbase = (lane >> 4) << 3;      // C/D rows this lane-half holds
  int khalf = (lane >> 4) << 4;      // A-frag K sub-offset (0 or 16)

  float bias3[NT], bias4[NT];
#pragma unroll
  for (int nt = 0; nt < NT; ++nt) {
    bias3[nt] = W3b[nt * 16 + col];
    bias4[nt] = W4b[nt * 16 + col];
  }

  int gwave = blockIdx.x * 8 + wave;
  int nwaves = gridDim.x * 8;

  for (int t = gwave; t < T; t += nwaves) {
    int e0 = lbound(seg_ids, E, t);
    int e1 = lbound(seg_ids, E, t + 1);
    int cnt = e1 - e0;

    float sumE[NT], sumS[NT];
#pragma unroll
    for (int nt = 0; nt < NT; ++nt) { sumE[nt] = 0.f; sumS[nt] = 0.f; }

    for (int cb = 0; cb < cnt; cb += 16) {
      int nvalid = min(16, cnt - cb);

      // ---- stage 16x384 edge-feature tile (f32 -> bf16) into LDS ----
      for (int p = lane; p < 16 * 192; p += 32) {
        int m = p / 192;
        int kk = (p - m * 192) * 2;
        float vx = 0.f, vy = 0.f;
        if (m < nvalid) {
          int eidx = e0 + cb + m;
          if (kk < 128) {
            float att = nbr_att[eidx];
            vx = fmaxf(att * W1w[kk]     + W1b[kk],     0.f);
            vy = fmaxf(att * W1w[kk + 1] + W1b[kk + 1], 0.f);
          } else if (kk < 256) {
            const float* r = ent + (size_t)nbr_ent[eidx] * HDIM + (kk - 128);
            vx = r[0]; vy = r[1];
          } else {
            const float* r = rel + (size_t)nbr_rel[eidx] * HDIM + (kk - 256);
            vx = r[0]; vy = r[1];
          }
        }
        unsigned int packed = f2bf(vx) | (f2bf(vy) << 16);
        ((unsigned int*)Atile)[(m * ASTRIDE + kk) >> 1] = packed;
      }
      asm volatile("s_wait_dscnt 0" ::: "memory");  // LDS stores visible to this wave

      // ---- load A fragments for all 12 K-steps (K=0..383) ----
      FragU a[KS3];
#pragma unroll
      for (int ks = 0; ks < KS3; ++ks) {
        int base = col * ASTRIDE + ks * 32 + khalf;   // 16B aligned
        a[ks].q[0] = *(const uint4*)(Atile + base);
        a[ks].q[1] = *(const uint4*)(Atile + base + 8);
      }

      // ---- e = relu(X @ W3^T + b3): 8 N-tiles x 12 K-steps ----
      // unroll 1: keep only one C-tile + transient B-frags live (no spills)
#pragma unroll 1
      for (int nt = 0; nt < NT; ++nt) {
        v8f c = {};
#pragma unroll
        for (int ks = 0; ks < KS3; ++ks) {
          FragU b;
          int fo = ((nt * KS3 + ks) * 32 + lane) * 16;
          b.q[0] = *(const uint4*)(sB3 + fo);
          b.q[1] = *(const uint4*)(sB3 + fo + 8);
          c = __builtin_amdgcn_wmma_f32_16x16x32_bf16(
              false, a[ks].v, false, b.v, (short)0, c, false, false);
        }
        float partial = 0.f;
#pragma unroll
        for (int r2 = 0; r2 < 8; ++r2)
          if (cb + mbase + r2 < cnt)
            partial += fmaxf(c[r2] + bias3[nt], 0.f);
        partial += __shfl_xor(partial, 16, 32);
        sumE[nt] += partial;
      }

      // ---- e_static = relu(X2 @ W4^T + b4): reuses A-frags K=128..383 ----
#pragma unroll 1
      for (int nt = 0; nt < NT; ++nt) {
        v8f c = {};
#pragma unroll
        for (int ks = 0; ks < KS4; ++ks) {
          FragU b;
          int fo = ((nt * KS4 + ks) * 32 + lane) * 16;
          b.q[0] = *(const uint4*)(sB4 + fo);
          b.q[1] = *(const uint4*)(sB4 + fo + 8);
          c = __builtin_amdgcn_wmma_f32_16x16x32_bf16(
              false, a[4 + ks].v, false, b.v, (short)0, c, false, false);
        }
        float partial = 0.f;
#pragma unroll
        for (int r2 = 0; r2 < 8; ++r2)
          if (cb + mbase + r2 < cnt)
            partial += fmaxf(c[r2] + bias4[nt], 0.f);
        partial += __shfl_xor(partial, 16, 32);
        sumS[nt] += partial;
      }
    }

    // ---- write segment means into output thirds ----
    float inv = 1.0f / (float)max(cnt, 1);
    if (lane < 16) {
      size_t base = (size_t)t * 384 + 256;
      size_t O2 = (size_t)T * 384;
#pragma unroll
      for (int nt = 0; nt < NT; ++nt) {
        out[base + nt * 16 + lane]      = sumS[nt] * inv;  // static mean -> s_embed
        out[O2 + base + nt * 16 + lane] = sumE[nt] * inv;  // mean       -> att_embed
      }
    }
  }
}

// ---------------------------------------------------------------------------
extern "C" void kernel_launch(void* const* d_in, const int* in_sizes, int n_in,
                              void* d_out, int out_size, void* d_ws, size_t ws_size,
                              hipStream_t stream) {
  const int*   nbr_ent  = (const int*)  d_in[0];
  const int*   nbr_rel  = (const int*)  d_in[1];
  const float* nbr_att  = (const float*)d_in[2];
  const int*   seg_ids  = (const int*)  d_in[3];
  const float* self_att = (const float*)d_in[4];
  const int*   s_ids    = (const int*)  d_in[5];
  const int*   r_ids    = (const int*)  d_in[6];
  const float* ent      = (const float*)d_in[7];
  const float* rel      = (const float*)d_in[8];
  const float* W1w      = (const float*)d_in[9];
  const float* W1b      = (const float*)d_in[10];
  const float* W3w      = (const float*)d_in[11];
  const float* W3b      = (const float*)d_in[12];
  const float* W4w      = (const float*)d_in[13];
  const float* W4b      = (const float*)d_in[14];

  float* out = (float*)d_out;
  int E = in_sizes[0];          // 262144
  int T = in_sizes[4];          // B*L = 10240
  unsigned short* Bpack = (unsigned short*)d_ws;   // 163840 B scratch

  // Allow >64KB dynamic LDS (320 KB per WGP on CDNA5).
  (void)hipFuncSetAttribute((const void*)seg_gemm_kernel,
                            hipFuncAttributeMaxDynamicSharedMemorySize,
                            SMEM_BYTES);

  pack_weights_kernel<<<(WPACK_USH + 255) / 256, 256, 0, stream>>>(W3w, W4w, Bpack);

  fill_outputs_kernel<<<(T * HDIM + 255) / 256, 256, 0, stream>>>(
      self_att, s_ids, r_ids, ent, rel, W1w, W1b, out, T);

  seg_gemm_kernel<<<320, 256, SMEM_BYTES, stream>>>(
      nbr_ent, nbr_rel, nbr_att, seg_ids, ent, rel,
      W1w, W1b, W3b, W4b, Bpack, out, E, T);
}
